// ChangeDetectorDoubleAttDyn_5076651344309
// MI455X (gfx1250) — compile-verified, matrix operand order checked by reading, not run on hardware
//
#include <hip/hip_runtime.h>
#include <math.h>

// ---------------- types for CDNA5 WMMA (bf16 stored as raw u16 bits) ----------------
typedef unsigned short u16;
typedef __attribute__((ext_vector_type(8)))  unsigned short u16x8;
typedef __attribute__((ext_vector_type(16))) unsigned short u16x16;
typedef __attribute__((ext_vector_type(16))) __bf16 v16bf;
typedef __attribute__((ext_vector_type(8)))  float v8f;

#define DEV static __device__ __forceinline__

#define BATCH 128
#define LTOK 256
#define LCAT 257
#define DMODEL 512
#define NHEAD 8
#define HDIM 64
#define NBP 96          // 81 rpe buckets padded to 96 (multiple of 32 for WMMA K)
#define LKP 288         // 257 keys padded to 288 (multiple of 32)
#define M1TOK (BATCH*LTOK)   // 32768
#define M2TOK (BATCH*LCAT)   // 32896

DEV u16 f2bf(float x){
  union { float f; unsigned u; } c; c.f = x;
  unsigned r = c.u + 0x7FFFu + ((c.u >> 16) & 1u);   // RNE f32->bf16
  return (u16)(r >> 16);
}
DEV v8f vzero8(){ v8f v;
#pragma unroll
  for (int i=0;i<8;i++) v[i]=0.f;
  return v; }

// 16-bit A/B fragment loader per ISA 16x32 layout:
// lane holds row (lane&15); K elements [kh..kh+7] in elems 0..7 and [kh+16..kh+23] in elems 8..15,
// kh = kbase + (lane>>4)*8.  Works for global and LDS (generic) pointers.
DEV v16bf load_frag(const u16* base, int ld, int row_lo, int row_limit, int kbase, int lane){
  u16x16 f;
  int r  = row_lo + (lane & 15);
  int kh = kbase + ((lane >> 4) << 3);
  if (r < row_limit){
    const u16* p = base + (long long)r * ld + kh;
    u16x8 c0 = *(const u16x8*)(p);
    u16x8 c1 = *(const u16x8*)(p + 16);
#pragma unroll
    for (int i=0;i<8;i++){ f[i]=c0[i]; f[i+8]=c1[i]; }
  } else {
#pragma unroll
    for (int i=0;i<16;i++) f[i]=0;
  }
  return __builtin_bit_cast(v16bf, f);
}

// unguarded fragment loader (all tile shapes exact -> no exec-mask branches)
DEV v16bf load_frag_nc(const u16* base, int ld, int row_lo, int kbase, int lane){
  u16x16 f;
  int r  = row_lo + (lane & 15);
  int kh = kbase + ((lane >> 4) << 3);
  const u16* p = base + (long long)r * ld + kh;
  u16x8 c0 = *(const u16x8*)(p);
  u16x8 c1 = *(const u16x8*)(p + 16);
#pragma unroll
  for (int i=0;i<8;i++){ f[i]=c0[i]; f[i+8]=c1[i]; }
  return __builtin_bit_cast(v16bf, f);
}

DEV v8f wmma_bf(v16bf a, v16bf b, v8f c){
  return __builtin_amdgcn_wmma_f32_16x16x32_bf16(false, a, false, b, (short)0, c, false, false);
}

// CDNA5 async global->LDS copy (16B per lane), tracked by ASYNCcnt
DEV void async_ld_b128(unsigned lds_off, const void* gptr){
  asm volatile("global_load_async_to_lds_b128 %0, %1, off"
               :: "v"(lds_off), "v"((unsigned long long)(size_t)gptr)
               : "memory");
}
DEV void wait_async_le2(){ asm volatile("s_wait_asynccnt 2" ::: "memory"); }
DEV void wait_async_0 (){ asm volatile("s_wait_asynccnt 0" ::: "memory"); }

// ---------------- generic tiled WMMA GEMM:  C[M,N] = A[M,K](lda) @ W[N,K]^T + bias ----------------
// block = 128 threads (4 wave32), block tile 64x64; W tile staged in LDS via
// double-buffered global_load_async_to_lds_b128 (shared by all 4 waves).
// REQUIRES: M%16==0 within grid coverage, N%64==0, K%32==0 (true for all call sites).
__global__ __launch_bounds__(128)
void k_gemm(const u16* __restrict__ A, int lda,
            const u16* __restrict__ W,
            const float* __restrict__ bias,
            float* __restrict__ Cf, int ldc,
            u16* __restrict__ Cb, int ldcb,
            int M, int N, int K)
{
  __shared__ __align__(16) u16 Wt[2][64*32];   // 2 x 4KB double buffer
  int t = threadIdx.x;
  int lane = t & 31, wave = t >> 5;
  int m0 = blockIdx.y * 64 + wave * 16;
  int n0 = blockIdx.x * 64;
  int KT = K >> 5;

  // stage one 64x32 W tile into LDS buffer `buf` (2 async b128 per thread)
  auto stage = [&](int k0, int buf){
#pragma unroll
    for (int i=0;i<2;i++){
      int e = t + 128*i;                 // 0..255 : 64 rows x 4 chunks of 8 bf16
      int row = e >> 2, ch = (e & 3) << 3;
      const u16* g = W + (long long)(n0 + row)*K + k0 + ch;
      unsigned lofs = (unsigned)(unsigned long long)(size_t)(&Wt[buf][row*32 + ch]);
      async_ld_b128(lofs, g);
    }
  };
  stage(0, 0);

  v8f acc[4];
#pragma unroll
  for (int j=0;j<4;j++) acc[j]=vzero8();

  for (int kt=0; kt<KT; kt++){
    if (kt+1 < KT){ stage((kt+1)<<5, (kt+1)&1); wait_async_le2(); }
    else          { wait_async_0(); }
    __syncthreads();
    v16bf a = load_frag_nc(A, lda, m0, kt<<5, lane);
    const u16* wt = &Wt[kt&1][0];
#pragma unroll
    for (int j=0;j<4;j++){
      v16bf b = load_frag_nc(wt, 32, 16*j, 0, lane);
      acc[j] = wmma_bf(a, b, acc[j]);
    }
    __syncthreads();                     // protect buffer reuse
  }
  int lh = lane >> 4, nl = lane & 15;
#pragma unroll
  for (int j=0;j<4;j++){
#pragma unroll
    for (int r=0;r<8;r++){
      int row = m0 + r + 8*lh, col = n0 + 16*j + nl;
      float v = acc[j][r] + (bias ? bias[col] : 0.f);
      if (Cf) Cf[(long long)row*ldc + col] = v;
      if (Cb) Cb[(long long)row*ldcb + col] = f2bf(v);
    }
  }
}

// ---------------- rpe projection:  Out[bh,l,0:96] = x_head(l,64) @ rpePad(96,64)^T ----------------
__global__ __launch_bounds__(32)
void k_rpe_proj(const u16* __restrict__ X, const u16* __restrict__ R,
                float* __restrict__ Out, int L)
{
  int lane = threadIdx.x;
  int bh = blockIdx.y, b = bh >> 3, h = bh & 7;
  int q0 = blockIdx.x * 16;
  const u16* xb = X + (long long)b * L * DMODEL + h * HDIM;
  v16bf a0 = load_frag_nc(xb, DMODEL, q0, 0, lane);
  v16bf a1 = load_frag_nc(xb, DMODEL, q0, 32, lane);
  int lh = lane >> 4, nl = lane & 15;
#pragma unroll
  for (int j=0;j<6;j++){
    v16bf b0 = load_frag_nc(R, HDIM, j*16, 0, lane);
    v8f acc = wmma_bf(a0, b0, vzero8());
    v16bf b1 = load_frag_nc(R, HDIM, j*16, 32, lane);
    acc = wmma_bf(a1, b1, acc);
#pragma unroll
    for (int r=0;r<8;r++){
      int row = q0 + r + 8*lh;
      Out[((long long)bh*L + row)*NBP + j*16 + nl] = acc[r];
    }
  }
}

// ---------------- per-head V transpose:  VT[bh][d][key] (zero-padded keys) ----------------
__global__ void k_vT(const u16* __restrict__ V, u16* __restrict__ VT, int Lk)
{
  int bh = blockIdx.y, b = bh >> 3, h = bh & 7;
  int idx = blockIdx.x * blockDim.x + threadIdx.x;
  if (idx >= HDIM * LKP) return;
  int d = idx / LKP, key = idx % LKP;
  u16 v = 0;
  if (key < Lk) v = V[((long long)b*Lk + key)*DMODEL + h*HDIM + d];
  VT[((long long)bh*HDIM + d)*LKP + key] = v;
}

// ---------------- fused attention: scores + rpe gather + softmax + P@V + bucket rpe_v ----------------
// RPE=true instantiation is the exact 256x256 self-attention (no bounds checks needed).
template<int NKT, bool RPE>
__global__ __launch_bounds__(32)
void k_attn(const u16* __restrict__ Q, const u16* __restrict__ Kmat,
            const u16* __restrict__ VT,
            const float* __restrict__ Qr, const float* __restrict__ Kr,
            const int* __restrict__ RBt, const u16* __restrict__ RVt,
            float* __restrict__ Ctx, int Lq, int Lk)
{
  __shared__ __align__(16) u16 Pt[16*LKP];
  __shared__ float wb[16*NBP];
  __shared__ __align__(16) u16 wbh[16*NBP];
  int lane = threadIdx.x;
  int bh = blockIdx.y, b = bh >> 3, h = bh & 7;
  int q0 = blockIdx.x * 16;
  const u16* qb = Q    + (long long)b*Lq*DMODEL + h*HDIM;
  const u16* kb = Kmat + (long long)b*Lk*DMODEL + h*HDIM;
  v16bf a0 = load_frag(qb, DMODEL, q0, Lq, 0, lane);
  v16bf a1 = load_frag(qb, DMODEL, q0, Lq, 32, lane);
  v8f sc[NKT];
#pragma unroll
  for (int kt=0; kt<NKT; kt++){
    v16bf b0 = load_frag(kb, DMODEL, kt*16, Lk, 0, lane);
    v8f t = wmma_bf(a0, b0, vzero8());
    v16bf b1 = load_frag(kb, DMODEL, kt*16, Lk, 32, lane);
    sc[kt] = wmma_bf(a1, b1, t);
  }
  { // zero P staging tile (and bucket weights)
    int* pz = (int*)Pt;
    for (int i=lane; i<16*LKP/2; i+=32) pz[i]=0;
    if (RPE) for (int i=lane; i<16*NBP; i+=32) wb[i]=0.f;
  }
  __syncthreads();
  int lh = lane >> 4, nl = lane & 15;
  float mx[8], sm[8];
#pragma unroll
  for (int r=0;r<8;r++) mx[r] = -1e30f;
#pragma unroll
  for (int kt=0; kt<NKT; kt++){
#pragma unroll
    for (int r=0;r<8;r++){
      int m=r+8*lh, qg=q0+m, key=kt*16+nl;
      float s = -1e30f;
      if (RPE || (qg < Lq && key < Lk)){
        s = sc[kt][r] * 0.125f;                       // 1/sqrt(64)
        if (RPE){
          int bkt = RBt[qg*LTOK + key];
          s += Qr[((long long)bh*Lq + qg)*NBP + bkt]
             + Kr[((long long)bh*Lk + key)*NBP + bkt];
        }
      }
      sc[kt][r]=s;
      mx[r]=fmaxf(mx[r], s);
    }
  }
#pragma unroll
  for (int r=0;r<8;r++){
    float v=mx[r];
    v=fmaxf(v,__shfl_xor(v,1,32)); v=fmaxf(v,__shfl_xor(v,2,32));
    v=fmaxf(v,__shfl_xor(v,4,32)); v=fmaxf(v,__shfl_xor(v,8,32));
    mx[r]=v; sm[r]=0.f;
  }
#pragma unroll
  for (int kt=0; kt<NKT; kt++){
#pragma unroll
    for (int r=0;r<8;r++){ float p=__expf(sc[kt][r]-mx[r]); sc[kt][r]=p; sm[r]+=p; }
  }
#pragma unroll
  for (int r=0;r<8;r++){
    float v=sm[r];
    v+=__shfl_xor(v,1,32); v+=__shfl_xor(v,2,32);
    v+=__shfl_xor(v,4,32); v+=__shfl_xor(v,8,32);
    sm[r]=1.f/v;
  }
#pragma unroll
  for (int kt=0; kt<NKT; kt++){
#pragma unroll
    for (int r=0;r<8;r++){
      int m=r+8*lh, qg=q0+m, key=kt*16+nl;
      float p = sc[kt][r]*sm[r];
      Pt[m*LKP + key] = f2bf(p);
      if (RPE){
        int bkt = RBt[qg*LTOK + key];
        atomicAdd(&wb[m*NBP + bkt], p);
      }
    }
  }
  __syncthreads();
  if (RPE){
    for (int i=lane;i<16*NBP;i+=32) wbh[i]=f2bf(wb[i]);
    __syncthreads();
  }
  v8f ca[4];
#pragma unroll
  for (int j=0;j<4;j++) ca[j]=vzero8();
  const u16* vtb = VT + (long long)bh*HDIM*LKP;
#pragma unroll
  for (int ks=0; ks<LKP/32; ks++){
    v16bf aP = load_frag_nc(Pt, LKP, 0, ks*32, lane);
#pragma unroll
    for (int j=0;j<4;j++){
      v16bf bV = load_frag_nc(vtb, LKP, j*16, ks*32, lane);
      ca[j] = wmma_bf(aP, bV, ca[j]);
    }
  }
  if (RPE){
#pragma unroll
    for (int ks=0; ks<NBP/32; ks++){
      v16bf aW = load_frag_nc(wbh, NBP, 0, ks*32, lane);
#pragma unroll
      for (int j=0;j<4;j++){
        v16bf bR = load_frag_nc(RVt, NBP, j*16, ks*32, lane);
        ca[j] = wmma_bf(aW, bR, ca[j]);
      }
    }
  }
#pragma unroll
  for (int j=0;j<4;j++){
#pragma unroll
    for (int r=0;r<8;r++){
      int m=r+8*lh, qg=q0+m;
      if (RPE || qg < Lq)
        Ctx[((long long)b*Lq + qg)*DMODEL + h*HDIM + j*16 + nl] = ca[j][r];
    }
  }
}

// ---------------- residual + layernorm over 512 ----------------
__global__ __launch_bounds__(256)
void k_ln(const float* __restrict__ X, const float* __restrict__ R,
          const float* __restrict__ g, const float* __restrict__ bb,
          float* __restrict__ Of, u16* __restrict__ Ob)
{
  long long row = blockIdx.x;
  int t = threadIdx.x;
  __shared__ float red[8];
  __shared__ float meanv, rstdv;
  float v0 = X[row*DMODEL+t]     + R[row*DMODEL+t];
  float v1 = X[row*DMODEL+256+t] + R[row*DMODEL+256+t];
  float s = v0 + v1;
  for (int m=1;m<32;m<<=1) s += __shfl_xor(s,m,32);
  if ((t&31)==0) red[t>>5]=s;
  __syncthreads();
  if (t==0){ float a=0; for(int i=0;i<8;i++)a+=red[i]; meanv=a/512.f; }
  __syncthreads();
  float mu=meanv;
  float d0=v0-mu, d1=v1-mu;
  float q=d0*d0+d1*d1;
  for (int m=1;m<32;m<<=1) q += __shfl_xor(q,m,32);
  if ((t&31)==0) red[t>>5]=q;
  __syncthreads();
  if (t==0){ float a=0; for(int i=0;i<8;i++)a+=red[i]; rstdv=rsqrtf(a/512.f+1e-6f); }
  __syncthreads();
  float rs=rstdv;
  float o0=d0*rs*g[t]+bb[t];
  float o1=d1*rs*g[t+256]+bb[t+256];
  Of[row*DMODEL+t]=o0; Of[row*DMODEL+256+t]=o1;
  if (Ob){ Ob[row*DMODEL+t]=f2bf(o0); Ob[row*DMODEL+256+t]=f2bf(o1); }
}

// ---------------- groupnorm(32 groups)+relu in place on token-major (B,256,128) ----------------
__global__ __launch_bounds__(256)
void k_gn(float* __restrict__ Y, const float* __restrict__ g, const float* __restrict__ bb)
{
  int b = blockIdx.x, grp = blockIdx.y, t = threadIdx.x;
  __shared__ float r1[8], r2[8];
  __shared__ float mv, rv;
  float vals[4]; float s=0.f, q=0.f;
#pragma unroll
  for (int c=0;c<4;c++){
    float v = Y[((long long)b*256+t)*128 + grp*4+c];
    vals[c]=v; s+=v; q+=v*v;
  }
  for (int m=1;m<32;m<<=1){ s+=__shfl_xor(s,m,32); q+=__shfl_xor(q,m,32); }
  if ((t&31)==0){ r1[t>>5]=s; r2[t>>5]=q; }
  __syncthreads();
  if (t==0){
    float a=0,bq=0; for(int i=0;i<8;i++){a+=r1[i]; bq+=r2[i];}
    float mu=a/1024.f; mv=mu; rv=rsqrtf(bq/1024.f - mu*mu + 1e-5f);
  }
  __syncthreads();
#pragma unroll
  for (int c=0;c<4;c++){
    int ch=grp*4+c;
    float o=(vals[c]-mv)*rv*g[ch]+bb[ch];
    Y[((long long)b*256+t)*128+ch]=fmaxf(o,0.f);
  }
}

// ---------------- small elementwise / setup kernels ----------------
__global__ void k_cast(const float* __restrict__ s, u16* __restrict__ d, long long n){
  long long i = blockIdx.x*(long long)blockDim.x + threadIdx.x;
  if (i<n) d[i]=f2bf(s[i]);
}
__global__ void k_zero(float* p, long long n){
  long long i = blockIdx.x*(long long)blockDim.x + threadIdx.x;
  if (i<n) p[i]=0.f;
}
__global__ void k_perm(const float* __restrict__ X, u16* __restrict__ O){
  long long i = blockIdx.x*(long long)blockDim.x + threadIdx.x;   // over B*256*1024
  if (i >= (long long)M1TOK*1024) return;
  int c = (int)(i & 1023); long long t = i >> 10;
  int hw = (int)(t & 255); int b = (int)(t >> 8);
  O[i] = f2bf(X[((long long)b*1024 + c)*256 + hw]);
}
DEV int pw_bucket(int d){
  float ad = fabsf((float)d);
  float v;
  if (ad <= 1.9f) v = (float)d;
  else {
    float far = 1.9f + logf(ad/1.9f)/logf(15.2f/1.9f)*(3.8f-1.9f);
    far = fminf(far, 3.8f);
    v = (d<0) ? -far : far;
  }
  return (int)rintf(v) + 4;
}
__global__ void k_rb(int* rb){
  int i = blockIdx.x, j = threadIdx.x;
  int dy = (i>>4)-(j>>4), dx = (i&15)-(j&15);
  rb[i*LTOK+j] = pw_bucket(dy)*9 + pw_bucket(dx);
}
__global__ void k_rpe_pad(const float* __restrict__ R, u16* __restrict__ O){
  int i = blockIdx.x*blockDim.x + threadIdx.x;   // 96*64
  if (i >= NBP*HDIM) return;
  int n=i>>6, d=i&63;
  O[i] = (n<81) ? f2bf(R[n*HDIM+d]) : (u16)0;
}
__global__ void k_rpe_vT(const float* __restrict__ R, u16* __restrict__ O){
  int i = blockIdx.x*blockDim.x + threadIdx.x;   // 64*96
  if (i >= HDIM*NBP) return;
  int d=i/NBP, n=i%NBP;
  O[i] = (n<81) ? f2bf(R[n*HDIM+d]) : (u16)0;
}
__global__ void k_cat(const float* __restrict__ cls, const float* __restrict__ T,
                      float* __restrict__ Of, u16* __restrict__ Ob){
  long long i = blockIdx.x*(long long)blockDim.x + threadIdx.x;   // B*257*512
  if (i >= (long long)M2TOK*DMODEL) return;
  int c = (int)(i & 511); long long r = i >> 9;
  int l = (int)(r % 257); int b = (int)(r / 257);
  float v = (l==0) ? cls[c] : T[((long long)b*256 + (l-1))*DMODEL + c];
  Of[i]=v; Ob[i]=f2bf(v);
}
__global__ void k_extract(const float* __restrict__ X, float* __restrict__ Of, u16* __restrict__ Ob){
  long long i = blockIdx.x*(long long)blockDim.x + threadIdx.x;   // B*256*512
  if (i >= (long long)M1TOK*DMODEL) return;
  int c = (int)(i & 511); long long r = i >> 9;
  int l = (int)(r & 255); int b = (int)(r >> 8);
  float v = X[((long long)b*257 + 1 + l)*DMODEL + c];
  Of[i]=v; Ob[i]=f2bf(v);
}
__global__ void k_diff(const float* __restrict__ A, const float* __restrict__ Bv,
                       float* __restrict__ Of, u16* __restrict__ Ob, long long n){
  long long i = blockIdx.x*(long long)blockDim.x + threadIdx.x;
  if (i>=n) return;
  float v = A[i]-Bv[i];
  Of[i]=v; Ob[i]=f2bf(v);
}
__global__ void k_place(const float* __restrict__ S, u16* __restrict__ D, long long rows, int off){
  long long i = blockIdx.x*(long long)blockDim.x + threadIdx.x;   // rows*512
  if (i >= rows*DMODEL) return;
  int c = (int)(i & 511); long long r = i >> 9;
  D[r*1536 + off + c] = f2bf(S[i]);
}
__global__ void k_gate(const float* __restrict__ G1, const float* __restrict__ G2,
                       const float* __restrict__ g2b,
                       const float* __restrict__ C1, const float* __restrict__ C2,
                       const float* __restrict__ c2b,
                       u16* __restrict__ D){
  long long i = blockIdx.x*(long long)blockDim.x + threadIdx.x;   // M1*512
  if (i >= (long long)M1TOK*DMODEL) return;
  int c = (int)(i & 511); long long r = i >> 9;
  float gv = G1[i]+G2[i]+g2b[c];
  float cv = C1[i]+C2[i]+c2b[c];
  float o = (1.f/(1.f+__expf(-gv))) * tanhf(cv);
  D[r*1536 + c] = f2bf(o);
}
__global__ void k_att(const float* __restrict__ Y, const float* __restrict__ attW,
                      const float* __restrict__ attb, float* __restrict__ Out){
  long long tok = blockIdx.x*(long long)blockDim.x + threadIdx.x;
  if (tok >= (long long)M1TOK) return;
  float s = attb[0];
  for (int c=0;c<128;c++) s += Y[tok*128+c]*attW[c];
  Out[tok] = 1.f/(1.f+__expf(-s));
}
__global__ __launch_bounds__(512)
void k_attended(const float* __restrict__ T, const float* __restrict__ A, float* __restrict__ Out){
  int b = blockIdx.x; int d = threadIdx.x;
  float s=0.f;
  for (int hw=0; hw<256; hw++) s += T[((long long)b*256+hw)*DMODEL+d]*A[b*256+hw];
  Out[b*DMODEL+d]=s;
}
__global__ __launch_bounds__(192)
void k_pred(const float* __restrict__ a1, const float* __restrict__ a2,
            const float* __restrict__ f1W, const float* __restrict__ f1b,
            float* __restrict__ Out){
  int b=blockIdx.x; int j=threadIdx.x>>5; int lane=threadIdx.x&31;
  float s=0.f;
  for (int d=lane; d<DMODEL; d+=32) s += (a2[b*DMODEL+d]-a1[b*DMODEL+d])*f1W[j*DMODEL+d];
  for (int m=1;m<32;m<<=1) s+=__shfl_xor(s,m,32);
  if (lane==0) Out[b*6+j]=s+f1b[j];
}
__global__ __launch_bounds__(128)
void k_loss(const float* __restrict__ c1, const float* __restrict__ c2, float* __restrict__ loss){
  int i=blockIdx.x, j=threadIdx.x;
  __shared__ float s1[128], s2[128];
  float d1=0.f,d2=0.f;
  for (int k=0;k<128;k++){ d1+=c1[i*128+k]*c2[j*128+k]; d2+=c2[i*128+k]*c1[j*128+k]; }
  s1[j]=2.f*d1; s2[j]=2.f*d2;
  __syncthreads();
  if (j==0){
    float m1=-1e30f,m2=-1e30f;
    for (int k=0;k<128;k++){ m1=fmaxf(m1,s1[k]); m2=fmaxf(m2,s2[k]); }
    float e1=0.f,e2=0.f;
    for (int k=0;k<128;k++){ e1+=__expf(s1[k]-m1); e2+=__expf(s2[k]-m2); }
    float lse1=m1+__logf(e1), lse2=m2+__logf(e2);
    float contrib = (lse1 - s1[i]) + (lse2 - s2[i]);
    atomicAdd(loss, contrib / (2.f*128.f));
  }
}

// ================================ host orchestration ================================
extern "C" void kernel_launch(void* const* d_in, const int* in_sizes, int n_in,
                              void* d_out, int out_size, void* d_ws, size_t ws_size,
                              hipStream_t stream)
{
  (void)in_sizes; (void)n_in; (void)out_size; (void)ws_size;
  const float* input_1=(const float*)d_in[0];  const float* input_2=(const float*)d_in[1];
  const float* W_img=(const float*)d_in[2];    const float* b_img=(const float*)d_in[3];
  const float* sWq=(const float*)d_in[4];  const float* sbq=(const float*)d_in[5];
  const float* sWk=(const float*)d_in[6];  const float* sbk=(const float*)d_in[7];
  const float* sWv=(const float*)d_in[8];  const float* sbv=(const float*)d_in[9];
  const float* sg =(const float*)d_in[10]; const float* sb =(const float*)d_in[11];
  const float* eWq=(const float*)d_in[12]; const float* ebq=(const float*)d_in[13];
  const float* eWk=(const float*)d_in[14]; const float* ebk=(const float*)d_in[15];
  const float* eWv=(const float*)d_in[16]; const float* ebv=(const float*)d_in[17];
  const float* eg =(const float*)d_in[18]; const float* eb =(const float*)d_in[19];
  const float* rpe_q=(const float*)d_in[20]; const float* rpe_k=(const float*)d_in[21];
  const float* rpe_v=(const float*)d_in[22];
  const float* bef_cls=(const float*)d_in[23]; const float* aft_cls=(const float*)d_in[24];
  const float* c1W=(const float*)d_in[25]; const float* c2W=(const float*)d_in[26];
  const float* c2b=(const float*)d_in[27];
  const float* g1W=(const float*)d_in[28]; const float* g2W=(const float*)d_in[29];
  const float* g2b=(const float*)d_in[30];
  const float* cpW=(const float*)d_in[31]; const float* cpb=(const float*)d_in[32];
  const float* embW=(const float*)d_in[33]; const float* embb=(const float*)d_in[34];
  const float* gng=(const float*)d_in[35]; const float* gnb=(const float*)d_in[36];
  const float* attW=(const float*)d_in[37]; const float* attb=(const float*)d_in[38];
  const float* f1W=(const float*)d_in[39]; const float* f1b=(const float*)d_in[40];

  char* ws = (char*)d_ws;
  size_t off = 0;
  auto alloc = [&](size_t bytes)->char*{
    char* p = ws + off; off += (bytes + 255) & ~(size_t)255; return p;
  };
  const long long TM1 = (long long)M1TOK, TM2 = (long long)M2TOK;

  u16* X1b = (u16*)alloc(TM1*1024*2);
  u16* X2b = (u16*)alloc(TM1*1024*2);
  u16* Wimgb=(u16*)alloc(512*1024*2);
  u16* sWqb=(u16*)alloc(512*512*2); u16* sWkb=(u16*)alloc(512*512*2); u16* sWvb=(u16*)alloc(512*512*2);
  u16* eWqb=(u16*)alloc(512*512*2); u16* eWkb=(u16*)alloc(512*512*2); u16* eWvb=(u16*)alloc(512*512*2);
  u16* c1Wb=(u16*)alloc(512*512*2); u16* c2Wb=(u16*)alloc(512*512*2);
  u16* g1Wb=(u16*)alloc(512*512*2); u16* g2Wb=(u16*)alloc(512*512*2);
  u16* cpWb=(u16*)alloc(128*512*2); u16* embWb=(u16*)alloc(128*1536*2);
  u16* rqp=(u16*)alloc(NBP*HDIM*2); u16* rkp=(u16*)alloc(NBP*HDIM*2); u16* rvT=(u16*)alloc(HDIM*NBP*2);
  int* rb = (int*)alloc(LTOK*LTOK*4);
  float* ibf=(float*)alloc(TM1*512*4); u16* ibb=(u16*)alloc(TM1*512*2);
  float* iaf=(float*)alloc(TM1*512*4); u16* iab=(u16*)alloc(TM1*512*2);
  u16* qb=(u16*)alloc(TM2*512*2); u16* kb=(u16*)alloc(TM2*512*2); u16* vb=(u16*)alloc(TM2*512*2);
  u16* vT=(u16*)alloc((long long)BATCH*NHEAD*HDIM*LKP*2);
  float* Qr=(float*)alloc((long long)BATCH*NHEAD*LTOK*NBP*4);
  float* Kr=(float*)alloc((long long)BATCH*NHEAD*LTOK*NBP*4);
  float* ctx=(float*)alloc(TM2*512*4);
  float* ibpost=(float*)alloc(TM1*512*4);
  float* iapost=(float*)alloc(TM1*512*4);
  float* catf_b=(float*)alloc(TM2*512*4); u16* catb_b=(u16*)alloc(TM2*512*2);
  float* catf_a=(float*)alloc(TM2*512*4); u16* catb_a=(u16*)alloc(TM2*512*2);
  float* ib2f=(float*)alloc(TM2*512*4); u16* ib2b=(u16*)alloc(TM2*512*2);
  float* ia2f=(float*)alloc(TM2*512*4); u16* ia2b=(u16*)alloc(TM2*512*2);
  float* tbf=(float*)alloc(TM1*512*4); u16* tbb=(u16*)alloc(TM1*512*2);
  float* taf=(float*)alloc(TM1*512*4); u16* tab=(u16*)alloc(TM1*512*2);
  float* difff=(float*)alloc(TM1*512*4); u16* diffb=(u16*)alloc(TM1*512*2);
  float* c1buf=(float*)alloc(128*128*4); float* c2buf=(float*)alloc(128*128*4);
  float* at1=(float*)alloc(128*512*4);   float* at2=(float*)alloc(128*512*4);
  // phase-based aliasing of dead buffers (all earlier users complete before reuse):
  float* G1 = Qr;                 // rpe tables dead after self-attn
  float* C1 = Kr;
  float* G2 = ctx;                // ctx dead after last LN
  float* C2 = (float*)qb;         // qkv bf16 pool (3 x 33.7MB contiguous) dead after cross-attn
  u16*   concat = X1b;            // input bf16 copies (134MB contiguous) dead after first GEMMs
  float* y = (float*)vT;          // vT dead after attention

  float* predP=(float*)d_out;
  float* attB = predP + 768;
  float* attA = predP + 768 + 32768;
  float* lossP= predP + 66304;

  auto gl = [](long long n, int tpb){ return dim3((unsigned)((n + tpb - 1)/tpb)); };
  auto cast = [&](const float* s, u16* d, long long n){
    k_cast<<<gl(n,256),256,0,stream>>>(s,d,n);
  };
  auto gemm = [&](const u16* A, int lda, const u16* W, const float* bias,
                  float* Cf, int ldc, u16* Cb, int ldcb, int M, int N, int K){
    dim3 g((N+63)/64, (M+63)/64);
    k_gemm<<<g,128,0,stream>>>(A,lda,W,bias,Cf,ldc,Cb,ldcb,M,N,K);
  };

  // ---- setup ----
  k_zero<<<1,32,0,stream>>>(lossP, 1);
  k_rb<<<LTOK,LTOK,0,stream>>>(rb);
  k_perm<<<gl(TM1*1024,256),256,0,stream>>>(input_1, X1b);
  k_perm<<<gl(TM1*1024,256),256,0,stream>>>(input_2, X2b);
  cast(W_img, Wimgb, 512*1024);
  cast(sWq,sWqb,512*512); cast(sWk,sWkb,512*512); cast(sWv,sWvb,512*512);
  cast(eWq,eWqb,512*512); cast(eWk,eWkb,512*512); cast(eWv,eWvb,512*512);
  cast(c1W,c1Wb,512*512); cast(c2W,c2Wb,512*512);
  cast(g1W,g1Wb,512*512); cast(g2W,g2Wb,512*512);
  cast(cpW,cpWb,128*512); cast(embW,embWb,128*1536);
  k_rpe_pad<<<24,256,0,stream>>>(rpe_q, rqp);
  k_rpe_pad<<<24,256,0,stream>>>(rpe_k, rkp);
  k_rpe_vT<<<24,256,0,stream>>>(rpe_v, rvT);

  // ---- image embedding GEMMs ----
  gemm(X1b,1024, Wimgb, b_img, ibf,512, ibb,512, M1TOK,512,1024);
  gemm(X2b,1024, Wimgb, b_img, iaf,512, iab,512, M1TOK,512,1024);

  // ---- self attention (per image, buffers reused) ----
  const u16* srcs[2] = { ibb, iab };
  const float* ress[2] = { ibf, iaf };
  float* posts[2] = { ibpost, iapost };
  for (int im=0; im<2; im++){
    gemm(srcs[im],512, sWqb, sbq, nullptr,0, qb,512, M1TOK,512,512);
    gemm(srcs[im],512, sWkb, sbk, nullptr,0, kb,512, M1TOK,512,512);
    gemm(srcs[im],512, sWvb, sbv, nullptr,0, vb,512, M1TOK,512,512);
    k_rpe_proj<<<dim3(16,BATCH*NHEAD),32,0,stream>>>(qb, rqp, Qr, LTOK);
    k_rpe_proj<<<dim3(16,BATCH*NHEAD),32,0,stream>>>(kb, rkp, Kr, LTOK);
    k_vT<<<dim3(72,BATCH*NHEAD),256,0,stream>>>(vb, vT, LTOK);
    k_attn<16,true><<<dim3(16,BATCH*NHEAD),32,0,stream>>>(qb,kb,vT,Qr,Kr,rb,rvT,ctx,LTOK,LTOK);
    k_ln<<<M1TOK,256,0,stream>>>(ctx, ress[im], sg, sb, posts[im], nullptr);
  }

  // ---- concat cls token ----
  k_cat<<<gl(TM2*512,256),256,0,stream>>>(bef_cls, ibpost, catf_b, catb_b);
  k_cat<<<gl(TM2*512,256),256,0,stream>>>(aft_cls, iapost, catf_a, catb_a);

  // ---- cross attention 1: q=ib_cat, k/v=ia_cat ----
  gemm(catb_b,512, eWqb, ebq, nullptr,0, qb,512, M2TOK,512,512);
  gemm(catb_a,512, eWkb, ebk, nullptr,0, kb,512, M2TOK,512,512);
  gemm(catb_a,512, eWvb, ebv, nullptr,0, vb,512, M2TOK,512,512);
  k_vT<<<dim3(72,BATCH*NHEAD),256,0,stream>>>(vb, vT, LCAT);
  k_attn<17,false><<<dim3(17,BATCH*NHEAD),32,0,stream>>>(qb,kb,vT,nullptr,nullptr,nullptr,nullptr,ctx,LCAT,LCAT);
  k_ln<<<M2TOK,256,0,stream>>>(ctx, catf_b, eg, eb, ib2f, ib2b);

  // ---- cross attention 2: q=ia_cat, k/v=ib2 ----
  gemm(catb_a,512, eWqb, ebq, nullptr,0, qb,512, M2TOK,512,512);
  gemm(ib2b,512,   eWkb, ebk, nullptr,0, kb,512, M2TOK,512,512);
  gemm(ib2b,512,   eWvb, ebv, nullptr,0, vb,512, M2TOK,512,512);
  k_vT<<<dim3(72,BATCH*NHEAD),256,0,stream>>>(vb, vT, LCAT);
  k_attn<17,false><<<dim3(17,BATCH*NHEAD),32,0,stream>>>(qb,kb,vT,nullptr,nullptr,nullptr,nullptr,ctx,LCAT,LCAT);
  k_ln<<<M2TOK,256,0,stream>>>(ctx, catf_a, eg, eb, ia2f, ia2b);

  // ---- contrastive loss from cls tokens (strided-A GEMM grabs row 0 of each batch) ----
  gemm(ib2b, LCAT*512, cpWb, cpb, c1buf,128, nullptr,0, 128,128,512);
  gemm(ia2b, LCAT*512, cpWb, cpb, c2buf,128, nullptr,0, 128,128,512);
  k_loss<<<128,128,0,stream>>>(c1buf, c2buf, lossP);

  // ---- tokens, diff, gated features ----
  k_extract<<<gl(TM1*512,256),256,0,stream>>>(ib2f, tbf, tbb);
  k_extract<<<gl(TM1*512,256),256,0,stream>>>(ia2f, taf, tab);
  k_diff<<<gl(TM1*512,256),256,0,stream>>>(taf, tbf, difff, diffb, TM1*512);
  gemm(diffb,512, g1Wb, nullptr, G1,512, nullptr,0, M1TOK,512,512);
  gemm(diffb,512, c1Wb, nullptr, C1,512, nullptr,0, M1TOK,512,512);

  for (int phase=0; phase<2; phase++){
    const u16* tk_b = (phase==0) ? tbb : tab;
    const float* tk_f = (phase==0) ? tbf : taf;
    float* attMap = (phase==0) ? attB : attA;
    float* att_out = (phase==0) ? at1 : at2;
    gemm(tk_b,512, g2Wb, nullptr, G2,512, nullptr,0, M1TOK,512,512);
    gemm(tk_b,512, c2Wb, nullptr, C2,512, nullptr,0, M1TOK,512,512);
    k_gate<<<gl(TM1*512,256),256,0,stream>>>(G1,G2,g2b,C1,C2,c2b,concat);    // cols 0..511
    k_place<<<gl(TM1*512,256),256,0,stream>>>(difff, concat, TM1, 512);      // cols 512..1023
    k_place<<<gl(TM1*512,256),256,0,stream>>>(tk_f,  concat, TM1, 1024);     // cols 1024..1535
    gemm(concat,1536, embWb, embb, y,128, nullptr,0, M1TOK,128,1536);
    k_gn<<<dim3(BATCH,32),256,0,stream>>>(y, gng, gnb);
    k_att<<<gl(TM1,256),256,0,stream>>>(y, attW, attb, attMap);
    k_attended<<<BATCH,512,0,stream>>>(tk_f, attMap, att_out);
  }
  k_pred<<<BATCH,192,0,stream>>>(at1, at2, f1W, f1b, predP);
}
